// TransformerBlock_33354716021099
// MI455X (gfx1250) — matrix operand order repaired and checked by reading, not compiled, not run on hardware
//
#include <hip/hip_runtime.h>
#include <hip/hip_bf16.h>
#include <math.h>

// ---------------------------------------------------------------------------
// Types for CDNA5 WMMA (wave32): v_wmma_f32_16x16x32_bf16
// ---------------------------------------------------------------------------
typedef __attribute__((ext_vector_type(16))) __bf16 v16bf;
typedef __attribute__((ext_vector_type(8)))  float  v8f;

union BF16x16 { v16bf v; uint4 q[2]; };
union F32x8   { v8f  v; float f[8]; };

#define WMMA_BF16(A, Bm, C) \
  __builtin_amdgcn_wmma_f32_16x16x32_bf16(false, (A), false, (Bm), (short)0, (C), false, false)

// One-b128 global->LDS async DMA (ASYNCcnt-tracked); offset applies to both sides.
#define ASYNC_LDS_B128(ldsoff_u32, gptr) \
  asm volatile("global_load_async_to_lds_b128 %0, %1, off" \
               :: "v"(ldsoff_u32), "v"(gptr) : "memory")
#define ASYNC_LDS_B128x2(ldsoff_u32, gptr) \
  asm volatile("global_load_async_to_lds_b128 %0, %1, off\n\t" \
               "global_load_async_to_lds_b128 %0, %1, off offset:16" \
               :: "v"(ldsoff_u32), "v"(gptr) : "memory")
#define WAIT_ASYNC0() asm volatile("s_wait_asynccnt 0x0" ::: "memory")

// Problem constants
constexpr int S_  = 512;
constexpr int BB_ = 8;
constexpr int D_  = 1024;
constexpr int H_  = 16;
constexpr int HD_ = 64;
constexpr int MR_ = S_ * BB_;                       // 4096 rows
constexpr size_t MD_ = (size_t)MR_ * D_;            // 4096*1024 elems
constexpr int SSZ = S_ * S_;

enum { EPI_F32 = 0, EPI_BF16 = 1, EPI_GELU_BF16 = 2, EPI_QKV = 3, EPI_QKV_VT = 4 };

// ---------------------------------------------------------------------------
// helpers
// ---------------------------------------------------------------------------
__device__ inline float wred_max16(float v) {
#pragma unroll
  for (int o = 1; o < 16; o <<= 1) v = fmaxf(v, __shfl_xor(v, o, 32));
  return v;
}
__device__ inline float wred_sum16(float v) {
#pragma unroll
  for (int o = 1; o < 16; o <<= 1) v += __shfl_xor(v, o, 32);
  return v;
}
__device__ inline unsigned lds_off(const void* p) {
  return (unsigned)(uintptr_t)p;  // generic->LDS: low 32 bits are the LDS offset
}

// ---------------------------------------------------------------------------
// f32 -> bf16 convert (row-major copy)
// ---------------------------------------------------------------------------
__global__ void cvt_bf16_kernel(const float* __restrict__ in, __bf16* __restrict__ out, int n) {
  int i = blockIdx.x * blockDim.x + threadIdx.x;
  if (i < n) out[i] = (__bf16)in[i];
}

// f32 [K][N] -> bf16 [N][K] (transpose; weights become k-contiguous B operands)
__global__ void cvt_bf16_t_kernel(const float* __restrict__ in, __bf16* __restrict__ out,
                                  int Kd, int Nd) {
  int i = blockIdx.x * blockDim.x + threadIdx.x;
  if (i < Kd * Nd) {
    int k = i / Nd, n = i % Nd;
    out[(size_t)n * Kd + k] = (__bf16)in[i];
  }
}

// ---------------------------------------------------------------------------
// Per-branch additive score-bias matrices: [4][512][512] f32
// ---------------------------------------------------------------------------
__global__ void build_sbias_kernel(const float* __restrict__ dist,
                                   const int* __restrict__ regA,
                                   const int* __restrict__ entA,
                                   const float* __restrict__ alpha_p,
                                   float* __restrict__ sb) {
  int i = blockIdx.x * blockDim.x + threadIdx.x;
  if (i < SSZ) {
    float al = alpha_p[0];
    sb[i]            = -al * dist[i];
    sb[SSZ + i]      = regA[i] ? 0.f : -1e9f;
    sb[2 * SSZ + i]  = entA[i] ? 0.f : -1e9f;
    sb[3 * SSZ + i]  = 0.f;
  }
}

// ---------------------------------------------------------------------------
// Tiled bf16 WMMA GEMM: C[M,N] = A[M,K] @ B[K,N] + bias[N].
// A row-major [M][K]; Bt pre-transposed [N][K]. Block tile 128x128, K-tile 32,
// 256 threads = 8 waves of 32x64. Double-buffered via async global->LDS DMA.
// ---------------------------------------------------------------------------
__global__ __launch_bounds__(256) void gemm_bf16_kernel(
    const __bf16* __restrict__ A, const __bf16* __restrict__ Bt,
    const float* __restrict__ bias, void* __restrict__ out,
    int M, int N, int K, int epi) {
  __shared__ __align__(16) __bf16 Ash[2][128 * 32];   // [m][k]
  __shared__ __align__(16) __bf16 Bts[2][128 * 32];   // [n][k]

  const int tid = threadIdx.x;
  const int wave = tid >> 5, lane = tid & 31;
  const int r16 = lane & 15, hi = lane >> 4;
  const int waveM = wave >> 1, waveN = wave & 1;
  const int nbase = blockIdx.x * 128;
  const int mbase = blockIdx.y * 128;

  F32x8 acc[2][4];
#pragma unroll
  for (int rm = 0; rm < 2; ++rm)
#pragma unroll
    for (int cn = 0; cn < 4; ++cn)
#pragma unroll
      for (int j = 0; j < 8; ++j) acc[rm][cn].f[j] = 0.f;

  const int nkt = K >> 5;
  const int lr = tid >> 1, lc = (tid & 1) * 16;  // tile loader coords (A and Bt identical)

  auto issue_tile = [&](int kt, int buf) {
    const int kb = kt << 5;
    {
      const __bf16* src = A + (size_t)(mbase + lr) * K + kb + lc;
      ASYNC_LDS_B128x2(lds_off(&Ash[buf][lr * 32 + lc]), src);
    }
    {
      const __bf16* src = Bt + (size_t)(nbase + lr) * K + kb + lc;
      ASYNC_LDS_B128x2(lds_off(&Bts[buf][lr * 32 + lc]), src);
    }
  };

  issue_tile(0, 0);
  WAIT_ASYNC0();
  __syncthreads();

  for (int kt = 0; kt < nkt; ++kt) {
    const int cur = kt & 1;
    if (kt + 1 < nkt) issue_tile(kt + 1, cur ^ 1);  // DMA next tile during compute

    BF16x16 af[2], bfr[4];
#pragma unroll
    for (int rm = 0; rm < 2; ++rm) {
      // A-layout: lane r16 = row; e=0..15 -> k=hi*16+e
      const uint4* ap = (const uint4*)(&Ash[cur][(waveM * 32 + rm * 16 + r16) * 32]);
      af[rm].q[0] = ap[2 * hi];
      af[rm].q[1] = ap[2 * hi + 1];
    }
#pragma unroll
    for (int cn = 0; cn < 4; ++cn) {
      // B-layout: lane r16 = col; e=0..15 -> k=hi*16+e
      const uint4* bp = (const uint4*)(&Bts[cur][(waveN * 64 + cn * 16 + r16) * 32 + hi * 16]);
      bfr[cn].q[0] = bp[0];
      bfr[cn].q[1] = bp[1];
    }
#pragma unroll
    for (int rm = 0; rm < 2; ++rm)
#pragma unroll
      for (int cn = 0; cn < 4; ++cn)
        acc[rm][cn].v = WMMA_BF16(af[rm].v, bfr[cn].v, acc[rm][cn].v);

    WAIT_ASYNC0();     // next tile fully landed in LDS
    __syncthreads();   // publish; all waves done reading cur
  }

  // epilogue: C layout -> VGPR j holds row j+hi*8, lane r16 = col
#pragma unroll
  for (int rm = 0; rm < 2; ++rm)
#pragma unroll
    for (int cn = 0; cn < 4; ++cn)
#pragma unroll
      for (int j = 0; j < 8; ++j) {
        const int row = mbase + waveM * 32 + rm * 16 + j + hi * 8;
        const int col = nbase + waveN * 64 + cn * 16 + r16;
        float v = acc[rm][cn].f[j] + bias[col];
        if (epi == EPI_F32) {
          ((float*)out)[(size_t)row * N + col] = v;
        } else if (epi == EPI_BF16) {
          ((__bf16*)out)[(size_t)row * N + col] = (__bf16)v;
        } else if (epi == EPI_GELU_BF16) {
          float u = 0.7978845608028654f * (v + 0.044715f * v * v * v);
          float g = 0.5f * v * (1.f + tanhf(u));
          ((__bf16*)out)[(size_t)row * N + col] = (__bf16)g;
        } else if (epi == EPI_QKV) {  // [S*B, D] -> [b][h][s][hd]
          const int s = row >> 3, bq = row & 7;
          const int hh = col >> 6, hd = col & 63;
          ((__bf16*)out)[((((size_t)bq * H_) + hh) * S_ + s) * HD_ + hd] = (__bf16)v;
        } else {                      // EPI_QKV_VT: [S*B, D] -> [b][h][hd][s]
          const int s = row >> 3, bq = row & 7;
          const int hh = col >> 6, hd = col & 63;
          ((__bf16*)out)[((((size_t)bq * H_) + hh) * HD_ + hd) * S_ + s] = (__bf16)v;
        }
      }
}

// ---------------------------------------------------------------------------
// Flash attention, one WG per (branch, b, h, quarter). 256 threads = 8 waves,
// each wave owns a 16-row Q tile; K [t][hd] and V^T [hd][t] tiles (128 t) in LDS,
// staged by async global->LDS DMA (V is already transposed in global memory).
// ---------------------------------------------------------------------------
__global__ __launch_bounds__(256) void attn_kernel(
    const __bf16* __restrict__ Qall, const __bf16* __restrict__ Kall,
    const __bf16* __restrict__ VTall, __bf16* __restrict__ Oall,
    const float* __restrict__ sbias) {
  const int wg = blockIdx.x;        // 2048
  const int quarter = wg & 3;
  const int combo = wg >> 2;        // 512 = branch*128 + b*16 + h
  const int branch = combo >> 7;
  const int bh = combo & 127;
  const int b_ = bh >> 4;
  const int h  = bh & 15;

  const size_t headoff = (size_t)branch * MD_ + (size_t)bh * S_ * HD_;
  const __bf16* Q  = Qall + headoff;   // [s][hd]
  const __bf16* K  = Kall + headoff;   // [s][hd]
  const __bf16* VT = VTall + headoff;  // [hd][s]
  const float* bias = sbias + (size_t)branch * SSZ;

  __shared__ __align__(16) __bf16 Ksh[128 * 64];     // [t_local][hd]   16KB
  __shared__ __align__(16) __bf16 VshT[64 * 128];    // [hd][t_local]   16KB
  __shared__ __align__(16) __bf16 Pst[8][16 * 64];   // per-wave probs  16KB

  const int tid = threadIdx.x;
  const int wave = tid >> 5, lane = tid & 31;
  const int r16 = lane & 15, hi = lane >> 4;
  const int rowbase = quarter * 128 + wave * 16;     // s of this wave's Q tile

  // Q fragments (A-matrix 16x64 -> two K=32 frags), straight from global
  BF16x16 qa0, qa1;
  {
    const uint4* qp = (const uint4*)(Q + (size_t)(rowbase + r16) * HD_);
    qa0.q[0] = qp[2 * hi];     qa0.q[1] = qp[2 * hi + 1];
    qa1.q[0] = qp[4 + 2 * hi]; qa1.q[1] = qp[5 + 2 * hi];
  }

  F32x8 oacc[4];
#pragma unroll
  for (int n = 0; n < 4; ++n)
#pragma unroll
    for (int j = 0; j < 8; ++j) oacc[n].f[j] = 0.f;
  float mrow[8], lrow[8];
#pragma unroll
  for (int j = 0; j < 8; ++j) { mrow[j] = -1e30f; lrow[j] = 0.f; }

  for (int tb = 0; tb < 4; ++tb) {
    __syncthreads();  // all reads of previous tile complete before DMA overwrites
    {  // async-DMA K tile (contiguous) and V^T tile (contiguous rows)
      const uint4* ks = (const uint4*)(K + (size_t)tb * 128 * HD_);
      uint4* kd = (uint4*)Ksh;
      for (int i = tid; i < 1024; i += 256) ASYNC_LDS_B128(lds_off(kd + i), ks + i);
      for (int i = tid; i < 1024; i += 256) {
        int hd = i >> 4, c = i & 15;  // 16 uint4 per hd row of 128 t values
        const uint4* vs = (const uint4*)(VT + (size_t)hd * S_ + tb * 128) + c;
        ASYNC_LDS_B128(lds_off(&VshT[hd * 128 + c * 8]), vs);
      }
    }
    WAIT_ASYNC0();
    __syncthreads();

    for (int cbi = 0; cbi < 2; ++cbi) {
      const int tloc = cbi * 64;
      const int tglob = tb * 128 + tloc;
      F32x8 st[4];
#pragma unroll
      for (int sub = 0; sub < 4; ++sub) {
        BF16x16 kb0, kb1;  // B = K^T: lane r16 = t col, e -> hd = hi*16+e (+32)
        const uint4* kp = (const uint4*)(Ksh + (tloc + sub * 16 + r16) * 64 + hi * 16);
        kb0.q[0] = kp[0]; kb0.q[1] = kp[1];
        const uint4* kp2 = (const uint4*)(Ksh + (tloc + sub * 16 + r16) * 64 + 32 + hi * 16);
        kb1.q[0] = kp2[0]; kb1.q[1] = kp2[1];
        v8f a = {};
        a = WMMA_BF16(qa0.v, kb0.v, a);
        a = WMMA_BF16(qa1.v, kb1.v, a);
        st[sub].v = a;
        const int col = tglob + sub * 16 + r16;
#pragma unroll
        for (int j = 0; j < 8; ++j) {
          const int row = rowbase + j + hi * 8;
          st[sub].f[j] = st[sub].f[j] * 0.125f + bias[row * S_ + col];
        }
      }
      // online softmax update (rows live across 16-lane halves)
      float corr[8];
#pragma unroll
      for (int j = 0; j < 8; ++j) {
        float bm = fmaxf(fmaxf(st[0].f[j], st[1].f[j]), fmaxf(st[2].f[j], st[3].f[j]));
        bm = wred_max16(bm);
        float mn = fmaxf(mrow[j], bm);
        corr[j] = expf(mrow[j] - mn);
        float ps = 0.f;
#pragma unroll
        for (int sub = 0; sub < 4; ++sub) {
          float p = expf(st[sub].f[j] - mn);
          st[sub].f[j] = p;
          ps += p;
        }
        ps = wred_sum16(ps);
        lrow[j] = lrow[j] * corr[j] + ps;
        mrow[j] = mn;
      }
#pragma unroll
      for (int n = 0; n < 4; ++n)
#pragma unroll
        for (int j = 0; j < 8; ++j) oacc[n].f[j] *= corr[j];

      // stage probs (C layout -> row-major bf16) in wave-private LDS
      __bf16* P = &Pst[wave][0];
#pragma unroll
      for (int sub = 0; sub < 4; ++sub)
#pragma unroll
        for (int j = 0; j < 8; ++j)
          P[(j + hi * 8) * 64 + sub * 16 + r16] = (__bf16)st[sub].f[j];

      // reload probs as A fragments (same-wave LDS ordering via dscnt)
      BF16x16 pa0, pa1;
      {
        const uint4* pp = (const uint4*)(P + r16 * 64);
        pa0.q[0] = pp[2 * hi];     pa0.q[1] = pp[2 * hi + 1];
        pa1.q[0] = pp[4 + 2 * hi]; pa1.q[1] = pp[5 + 2 * hi];
      }
      // O += P @ V  (B from V^T: lane r16 = hd col, e -> t = tloc + hi*16 + e (+32))
#pragma unroll
      for (int n = 0; n < 4; ++n) {
        BF16x16 vb0, vb1;
        const uint4* vp = (const uint4*)(VshT + (n * 16 + r16) * 128 + tloc + hi * 16);
        vb0.q[0] = vp[0]; vb0.q[1] = vp[1];
        const uint4* vp2 = (const uint4*)(VshT + (n * 16 + r16) * 128 + tloc + 32 + hi * 16);
        vb1.q[0] = vp2[0]; vb1.q[1] = vp2[1];
        oacc[n].v = WMMA_BF16(pa0.v, vb0.v, oacc[n].v);
        oacc[n].v = WMMA_BF16(pa1.v, vb1.v, oacc[n].v);
      }
    }
  }

  // normalize + write O back as [S*B, D] bf16 for the O-projection GEMM
  float inv[8];
#pragma unroll
  for (int j = 0; j < 8; ++j) inv[j] = 1.f / lrow[j];
  __bf16* Ob = Oall + (size_t)branch * MD_;
#pragma unroll
  for (int n = 0; n < 4; ++n)
#pragma unroll
    for (int j = 0; j < 8; ++j) {
      const int s = rowbase + j + hi * 8;
      const int col = h * 64 + n * 16 + r16;
      Ob[(size_t)(s * BB_ + b_) * D_ + col] = (__bf16)(oacc[n].f[j] * inv[j]);
    }
}

// ---------------------------------------------------------------------------
// Gated fusion: gates = softmax(cat @ Wg + bg); fused = sum_k branch_k*g_k (bf16)
// ---------------------------------------------------------------------------
__global__ __launch_bounds__(256) void gates_fuse_kernel(
    const float* __restrict__ brout, const float* __restrict__ Wg,
    const float* __restrict__ bg, __bf16* __restrict__ fusedb) {
  const int row = blockIdx.x;
  const int tid = threadIdx.x;
  float a0 = 0, a1 = 0, a2 = 0, a3 = 0;
  for (int d = tid; d < D_; d += 256) {
#pragma unroll
    for (int kbr = 0; kbr < 4; ++kbr) {
      float v = brout[(size_t)kbr * MD_ + (size_t)row * D_ + d];
      const float* w = Wg + ((size_t)kbr * D_ + d) * 4;
      a0 += v * w[0]; a1 += v * w[1]; a2 += v * w[2]; a3 += v * w[3];
    }
  }
  __shared__ float sred[4][256];
  sred[0][tid] = a0; sred[1][tid] = a1; sred[2][tid] = a2; sred[3][tid] = a3;
  __syncthreads();
  for (int off = 128; off > 0; off >>= 1) {
    if (tid < off)
      for (int c = 0; c < 4; ++c) sred[c][tid] += sred[c][tid + off];
    __syncthreads();
  }
  __shared__ float g4[4];
  if (tid == 0) {
    float z[4];
    for (int c = 0; c < 4; ++c) z[c] = sred[c][0] + bg[c];
    float mx = fmaxf(fmaxf(z[0], z[1]), fmaxf(z[2], z[3]));
    float e[4], sum = 0.f;
    for (int c = 0; c < 4; ++c) { e[c] = expf(z[c] - mx); sum += e[c]; }
    for (int c = 0; c < 4; ++c) g4[c] = e[c] / sum;
  }
  __syncthreads();
  const float g0 = g4[0], g1 = g4[1], g2 = g4[2], g3 = g4[3];
  for (int d = tid; d < D_; d += 256) {
    const size_t base = (size_t)row * D_ + d;
    float f = g0 * brout[base] + g1 * brout[MD_ + base] +
              g2 * brout[2 * MD_ + base] + g3 * brout[3 * MD_ + base];
    fusedb[base] = (__bf16)f;
  }
}

// ---------------------------------------------------------------------------
// out = LayerNorm(x + y); optionally also emit bf16 copy
// ---------------------------------------------------------------------------
__global__ __launch_bounds__(256) void add_ln_kernel(
    const float* __restrict__ x, const float* __restrict__ y,
    const float* __restrict__ gam, const float* __restrict__ bet,
    float* __restrict__ outf, __bf16* __restrict__ outb) {
  const int row = blockIdx.x;
  const int tid = threadIdx.x;
  const size_t base = (size_t)row * D_;
  float s = 0.f, s2 = 0.f;
  for (int d = tid; d < D_; d += 256) {
    float v = x[base + d] + y[base + d];
    s += v; s2 += v * v;
  }
  __shared__ float r1[256], r2[256];
  r1[tid] = s; r2[tid] = s2;
  __syncthreads();
  for (int off = 128; off > 0; off >>= 1) {
    if (tid < off) { r1[tid] += r1[tid + off]; r2[tid] += r2[tid + off]; }
    __syncthreads();
  }
  __shared__ float smean, srstd;
  if (tid == 0) {
    float mean = r1[0] * (1.f / D_);
    float var = r2[0] * (1.f / D_) - mean * mean;
    smean = mean;
    srstd = rsqrtf(var + 1e-5f);
  }
  __syncthreads();
  const float mean = smean, rstd = srstd;
  for (int d = tid; d < D_; d += 256) {
    float v = (x[base + d] + y[base + d] - mean) * rstd * gam[d] + bet[d];
    outf[base + d] = v;
    if (outb) outb[base + d] = (__bf16)v;
  }
}

// ---------------------------------------------------------------------------
// Host orchestration
// ---------------------------------------------------------------------------
extern "C" void kernel_launch(void* const* d_in, const int* in_sizes, int n_in,
                              void* d_out, int out_size, void* d_ws, size_t ws_size,
                              hipStream_t stream) {
  (void)in_sizes; (void)n_in; (void)out_size; (void)ws_size;
  const float* x     = (const float*)d_in[0];
  const float* dist  = (const float*)d_in[1];
  const int*   regA  = (const int*)d_in[2];
  const int*   entA  = (const int*)d_in[3];
  const float* attnW = (const float*)d_in[4];   // [4][4][D][D]
  const float* attnB = (const float*)d_in[5];   // [4][4][D]
  const float* alpha = (const float*)d_in[6];
  const float* Wg    = (const float*)d_in[7];
  const float* bg    = (const float*)d_in[8];
  const float* Wf    = (const float*)d_in[9];
  const float* bfp   = (const float*)d_in[10];
  const float* W1    = (const float*)d_in[11];
  const float* b1    = (const float*)d_in[12];
  const float* W2    = (const float*)d_in[13];
  const float* b2    = (const float*)d_in[14];
  const float* g1    = (const float*)d_in[15];
  const float* be1   = (const float*)d_in[16];
  const float* g2    = (const float*)d_in[17];
  const float* be2   = (const float*)d_in[18];

  char* ws = (char*)d_ws;
  size_t cur = 0;
  auto alloc = [&](size_t bytes) -> void* {
    void* p = ws + cur;
    cur += (bytes + 255) & ~(size_t)255;
    return p;
  };

  __bf16* xb    = (__bf16*)alloc(MD_ * 2);
  __bf16* wAtt  = (__bf16*)alloc((size_t)16 * D_ * D_ * 2);   // each [N][K] transposed
  __bf16* wFf   = (__bf16*)alloc((size_t)D_ * D_ * 2);
  __bf16* w1b   = (__bf16*)alloc((size_t)D_ * 4 * D_ * 2);
  __bf16* w2b   = (__bf16*)alloc((size_t)4 * D_ * D_ * 2);
  __bf16* Qall  = (__bf16*)alloc(4 * MD_ * 2);
  __bf16* Kall  = (__bf16*)alloc(4 * MD_ * 2);
  __bf16* VTall = (__bf16*)alloc(4 * MD_ * 2);                // [br][b][h][hd][s]
  __bf16* Oatt  = (__bf16*)alloc(4 * MD_ * 2);
  float*  brout = (float*)alloc(4 * MD_ * 4);
  float*  sbias = (float*)alloc((size_t)4 * SSZ * 4);
  __bf16* fusedb= (__bf16*)alloc(MD_ * 2);
  float*  ybuf  = (float*)alloc(MD_ * 4);
  float*  x1    = (float*)alloc(MD_ * 4);
  __bf16* x1b   = (__bf16*)alloc(MD_ * 2);
  __bf16* h1b   = (__bf16*)alloc((size_t)MR_ * 4 * D_ * 2);
  float*  h2    = (float*)alloc(MD_ * 4);

  auto cvt = [&](const float* in, __bf16* outp, size_t n) {
    cvt_bf16_kernel<<<dim3((unsigned)((n + 255) / 256)), dim3(256), 0, stream>>>(in, outp, (int)n);
  };
  auto cvt_t = [&](const float* in, __bf16* outp, int Kd, int Nd) {
    size_t n = (size_t)Kd * Nd;
    cvt_bf16_t_kernel<<<dim3((unsigned)((n + 255) / 256)), dim3(256), 0, stream>>>(in, outp, Kd, Nd);
  };
  cvt(x, xb, MD_);
  for (int m = 0; m < 16; ++m)
    cvt_t(attnW + (size_t)m * D_ * D_, wAtt + (size_t)m * D_ * D_, D_, D_);
  cvt_t(Wf, wFf, D_, D_);
  cvt_t(W1, w1b, D_, 4 * D_);
  cvt_t(W2, w2b, 4 * D_, D_);

  build_sbias_kernel<<<dim3((SSZ + 255) / 256), dim3(256), 0, stream>>>(dist, regA, entA, alpha, sbias);

  auto gemm = [&](const __bf16* Ap, const __bf16* Btp, const float* biasp,
                  void* outp, int Mm, int Nn, int Kk, int epi) {
    gemm_bf16_kernel<<<dim3(Nn / 128, Mm / 128), dim3(256), 0, stream>>>(
        Ap, Btp, biasp, outp, Mm, Nn, Kk, epi);
  };

  // Q/K/V projections (scatter into [b][h][s][hd]; V transposed to [b][h][hd][s])
  for (int br = 0; br < 4; ++br) {
    gemm(xb, wAtt + ((size_t)br * 4 + 0) * D_ * D_, attnB + ((size_t)br * 4 + 0) * D_,
         Qall + (size_t)br * MD_, MR_, D_, D_, EPI_QKV);
    gemm(xb, wAtt + ((size_t)br * 4 + 1) * D_ * D_, attnB + ((size_t)br * 4 + 1) * D_,
         Kall + (size_t)br * MD_, MR_, D_, D_, EPI_QKV);
    gemm(xb, wAtt + ((size_t)br * 4 + 2) * D_ * D_, attnB + ((size_t)br * 4 + 2) * D_,
         VTall + (size_t)br * MD_, MR_, D_, D_, EPI_QKV_VT);
  }

  attn_kernel<<<dim3(2048), dim3(256), 0, stream>>>(Qall, Kall, VTall, Oatt, sbias);

  // O projections -> f32 branch outputs
  for (int br = 0; br < 4; ++br)
    gemm(Oatt + (size_t)br * MD_, wAtt + ((size_t)br * 4 + 3) * D_ * D_,
         attnB + ((size_t)br * 4 + 3) * D_, brout + (size_t)br * MD_,
         MR_, D_, D_, EPI_F32);

  gates_fuse_kernel<<<dim3(MR_), dim3(256), 0, stream>>>(brout, Wg, bg, fusedb);
  gemm(fusedb, wFf, bfp, ybuf, MR_, D_, D_, EPI_F32);
  add_ln_kernel<<<dim3(MR_), dim3(256), 0, stream>>>(x, ybuf, g1, be1, x1, x1b);

  gemm(x1b, w1b, b1, h1b, MR_, 4 * D_, D_, EPI_GELU_BF16);
  gemm(h1b, w2b, b2, h2, MR_, D_, 4 * D_, EPI_F32);
  add_ln_kernel<<<dim3(MR_), dim3(256), 0, stream>>>(x1, h2, g2, be2, (float*)d_out, (__bf16*)nullptr);
}
